// ButterflyMlp_3547642986663
// MI455X (gfx1250) — matrix operand order repaired and checked by reading, not compiled, not run on hardware
//
#include <hip/hip_runtime.h>

typedef __bf16 bf16_t;
typedef __attribute__((ext_vector_type(16))) __bf16 v16bf;
typedef __attribute__((ext_vector_type(8)))  __bf16 v8bf;
typedef __attribute__((ext_vector_type(8)))  float  v8f;
typedef __attribute__((ext_vector_type(4)))  float  v4f;

#define MT      32          // batch rows per workgroup
#define N1      784
#define KP      800         // K padded: 784 -> 800 = 25 * 32
#define N2      128
#define K3      128
#define NT1     (N1 / 16)   // 49 N-tiles in layer 1
#define NWAVES  8

// ---------------------------------------------------------------------------
// A-fragment (16x32 bf16): lane L<16 holds row M=L, K = k0+{0..7} in elems 0-7
// and K = k0+{16..23} in elems 8-15; lane L>=16 holds K = k0+{8..15, 24..31}.
// Caller passes p = rowBase + k0 + (lane>=16 ? 8 : 0).
__device__ __forceinline__ v16bf load_frag_A(const bf16_t* p) {
    v8bf lo = *(const v8bf*)(p);
    v8bf hi = *(const v8bf*)(p + 16);
    v16bf r;
#pragma unroll
    for (int i = 0; i < 8; ++i) { r[i] = lo[i]; r[i + 8] = hi[i]; }
    return r;
}

// B-fragment (32x16 bf16): lane L<16 is column N=L holding K = k0+{0..7, 16..23};
// lane L>=16 is column N=L-16 holding K = k0+{8..15, 24..31}.
// Caller passes p = colRowBase + k0 + (lane>=16 ? 8 : 0).
__device__ __forceinline__ v16bf load_frag_B(const bf16_t* p) {
    v8bf lo = *(const v8bf*)(p);
    v8bf hi = *(const v8bf*)(p + 16);
    v16bf r;
#pragma unroll
    for (int i = 0; i < 8; ++i) { r[i] = lo[i]; r[i + 8] = hi[i]; }
    return r;
}

// ---------------------------------------------------------------------------
// Bake (W * mask) into bf16, zero-padding K to kPad and rows to outPad.
__global__ __launch_bounds__(256) void prep_weights_kernel(
    const float* __restrict__ W, const unsigned char* __restrict__ mask,
    bf16_t* __restrict__ out, int realOut, int inF, int kPad, int outPad)
{
    int idx = blockIdx.x * 256 + threadIdx.x;
    int total = outPad * kPad;
    if (idx >= total) return;
    int o = idx / kPad;
    int k = idx - o * kPad;
    float v = 0.0f;
    if (o < realOut && k < inF && mask[(size_t)o * inF + k] != 0)
        v = W[(size_t)o * inF + k];
    out[idx] = (bf16_t)v;
}

// ---------------------------------------------------------------------------
__global__ __launch_bounds__(256) void butterfly_mlp_kernel(
    const float* __restrict__ x,
    const float* __restrict__ b1, const float* __restrict__ b2,
    const float* __restrict__ b3,
    const bf16_t* __restrict__ W1b,   // [784][800] bf16, masked, padded
    const bf16_t* __restrict__ W2b,   // [128][800]
    const bf16_t* __restrict__ W3b,   // [16][128] (rows 10..15 zero)
    float* __restrict__ out)
{
    __shared__ bf16_t Xs [MT * KP];   // 51200 B
    __shared__ bf16_t H1s[MT * KP];   // 51200 B
    __shared__ bf16_t H2s[MT * N2];   //  8192 B
    __shared__ bf16_t W3s[16 * K3];   //  4096 B (async-filled)
    __shared__ float  OutS[MT * 16];  //  2048 B

    const int tid  = threadIdx.x;
    const int wave = tid >> 5;
    const int lane = tid & 31;
    const int hl   = lane >> 4;    // 0 = low half, 1 = high half
    const int l16  = lane & 15;
    const size_t row0 = (size_t)blockIdx.x * MT;

    // ---- Async: W3 tile (4KB) global -> LDS, one b128 per thread.  Hidden
    // behind GEMM1+GEMM2; waited on before the pre-GEMM3 barrier. ----------
    {
        unsigned      ldsoff = (unsigned)(uintptr_t)(&W3s[tid * 8]);
        const bf16_t* gsrc   = W3b + tid * 8;
        asm volatile("global_load_async_to_lds_b128 %0, %1, off"
                     :: "v"(ldsoff), "v"(gsrc) : "memory");
    }

    // ---- Stage 0: x tile -> bf16 LDS (f32x4 NT global loads), zero K-pad --
    for (int i = tid; i < MT * (N1 / 4); i += 256) {
        int r  = i / (N1 / 4);
        int c4 = (i - r * (N1 / 4)) * 4;
        const v4f v = __builtin_nontemporal_load(
            ((const v4f*)(x + (row0 + r) * N1)) + (c4 >> 2));
        bf16_t* dst = &Xs[r * KP + c4];
        dst[0] = (bf16_t)v.x; dst[1] = (bf16_t)v.y;
        dst[2] = (bf16_t)v.z; dst[3] = (bf16_t)v.w;
    }
    for (int i = tid; i < MT * (KP - N1); i += 256) {
        int r = i / (KP - N1);
        int c = N1 + (i - r * (KP - N1));
        Xs [r * KP + c] = (bf16_t)0.0f;
        H1s[r * KP + c] = (bf16_t)0.0f;   // pad h1 too (read by GEMM2)
    }
    __syncthreads();

    // ---- GEMM1: h1 = relu(X @ W1m^T + b1), N=784, K=800 -------------------
    for (int nt = wave; nt < NT1; nt += NWAVES) {
        const int n0 = nt * 16;
        // Speculative prefetch of the next N-tile's weight slab (wave-uniform)
        if (nt + NWAVES < NT1)
            __builtin_prefetch(W1b + (size_t)((nt + NWAVES) * 16 + l16) * KP, 0, 0);
        const float bv = b1[n0 + l16];
        v8f acc0, acc1;
#pragma unroll
        for (int i = 0; i < 8; ++i) { acc0[i] = bv; acc1[i] = bv; }
        const bf16_t* brow  = W1b + (size_t)(n0 + l16) * KP + hl * 8;
        const bf16_t* arow0 = Xs  + (0 * 16 + l16) * KP + hl * 8;
        const bf16_t* arow1 = Xs  + (1 * 16 + l16) * KP + hl * 8;
        for (int k0 = 0; k0 < KP; k0 += 32) {
            v16bf bf = load_frag_B(brow  + k0);
            v16bf a0 = load_frag_A(arow0 + k0);
            v16bf a1 = load_frag_A(arow1 + k0);
            acc0 = __builtin_amdgcn_wmma_f32_16x16x32_bf16(
                       false, a0, false, bf, (short)0, acc0, false, false);
            acc1 = __builtin_amdgcn_wmma_f32_16x16x32_bf16(
                       false, a1, false, bf, (short)0, acc1, false, false);
        }
#pragma unroll
        for (int r = 0; r < 8; ++r) {
            const int rr = r + hl * 8;
            H1s[(0 * 16 + rr) * KP + n0 + l16] = (bf16_t)fmaxf(acc0[r], 0.0f);
            H1s[(1 * 16 + rr) * KP + n0 + l16] = (bf16_t)fmaxf(acc1[r], 0.0f);
        }
    }
    __syncthreads();

    // ---- GEMM2: h2 = relu(h1 @ W2m^T + b2), N=128 (1 N-tile/wave), K=800 --
    {
        const int n0 = wave * 16;
        const float bv = b2[n0 + l16];
        v8f acc0, acc1;
#pragma unroll
        for (int i = 0; i < 8; ++i) { acc0[i] = bv; acc1[i] = bv; }
        const bf16_t* brow  = W2b + (size_t)(n0 + l16) * KP + hl * 8;
        const bf16_t* arow0 = H1s + (0 * 16 + l16) * KP + hl * 8;
        const bf16_t* arow1 = H1s + (1 * 16 + l16) * KP + hl * 8;
        for (int k0 = 0; k0 < KP; k0 += 32) {
            v16bf bf = load_frag_B(brow  + k0);
            v16bf a0 = load_frag_A(arow0 + k0);
            v16bf a1 = load_frag_A(arow1 + k0);
            acc0 = __builtin_amdgcn_wmma_f32_16x16x32_bf16(
                       false, a0, false, bf, (short)0, acc0, false, false);
            acc1 = __builtin_amdgcn_wmma_f32_16x16x32_bf16(
                       false, a1, false, bf, (short)0, acc1, false, false);
        }
#pragma unroll
        for (int r = 0; r < 8; ++r) {
            const int rr = r + hl * 8;
            H2s[(0 * 16 + rr) * N2 + n0 + l16] = (bf16_t)fmaxf(acc0[r], 0.0f);
            H2s[(1 * 16 + rr) * N2 + n0 + l16] = (bf16_t)fmaxf(acc1[r], 0.0f);
        }
    }
    // W3 async copy must have landed in LDS before the barrier below.
    asm volatile("s_wait_asynccnt 0x0" ::: "memory");
    __syncthreads();

    // ---- GEMM3: logits = h2 @ W3m^T (16 padded cols), K=128, B from LDS ---
    if (wave < 2) {                         // wave-uniform branch, EXEC all-1s
        v8f acc;
#pragma unroll
        for (int i = 0; i < 8; ++i) acc[i] = 0.0f;
        const bf16_t* brow = W3s + l16 * K3 + hl * 8;
        const bf16_t* arow = H2s + (wave * 16 + l16) * N2 + hl * 8;
#pragma unroll
        for (int k0 = 0; k0 < K3; k0 += 32) {
            v16bf bf = load_frag_B(brow + k0);
            v16bf a  = load_frag_A(arow + k0);
            acc = __builtin_amdgcn_wmma_f32_16x16x32_bf16(
                      false, a, false, bf, (short)0, acc, false, false);
        }
#pragma unroll
        for (int r = 0; r < 8; ++r)
            OutS[(wave * 16 + r + hl * 8) * 16 + l16] = acc[r];
    }
    __syncthreads();

    // ---- Finalize: bias + log_softmax over 10 logits, one row per lane ----
    if (tid < MT) {                         // exactly wave 0, all lanes active
        const int row = tid;
        float v[10];
        float m = -3.0e38f;
#pragma unroll
        for (int j = 0; j < 10; ++j) {
            v[j] = OutS[row * 16 + j] + b3[j];
            m = fmaxf(m, v[j]);
        }
        float s = 0.0f;
#pragma unroll
        for (int j = 0; j < 10; ++j) s += __expf(v[j] - m);
        const float lse = m + __logf(s);
        float* o = out + (row0 + row) * 10;
#pragma unroll
        for (int j = 0; j < 10; ++j)
            __builtin_nontemporal_store(v[j] - lse, o + j);
    }
}

// ---------------------------------------------------------------------------
extern "C" void kernel_launch(void* const* d_in, const int* in_sizes, int n_in,
                              void* d_out, int out_size, void* d_ws, size_t ws_size,
                              hipStream_t stream) {
    (void)n_in; (void)out_size; (void)ws_size;

    const float* x  = (const float*)d_in[0];
    const float* W1 = (const float*)d_in[1];
    const float* b1 = (const float*)d_in[2];
    const float* W2 = (const float*)d_in[3];
    const float* b2 = (const float*)d_in[4];
    const float* W3 = (const float*)d_in[5];
    const float* b3 = (const float*)d_in[6];
    const unsigned char* m1 = (const unsigned char*)d_in[7];
    const unsigned char* m2 = (const unsigned char*)d_in[8];
    const unsigned char* m3 = (const unsigned char*)d_in[9];
    float* out = (float*)d_out;

    // Workspace layout (bf16, 256B-aligned offsets)
    char* ws = (char*)d_ws;
    bf16_t* W1b = (bf16_t*)(ws);                               // 784*800*2 = 1254400 B
    bf16_t* W2b = (bf16_t*)(ws + 1254400);                     // 128*800*2 =  204800 B
    bf16_t* W3b = (bf16_t*)(ws + 1254400 + 204800);            //  16*128*2 =    4096 B

    // Bake masked weights to bf16 (padded)
    prep_weights_kernel<<<(784 * KP + 255) / 256, 256, 0, stream>>>(
        W1, m1, W1b, /*realOut=*/784, /*inF=*/784, /*kPad=*/KP, /*outPad=*/784);
    prep_weights_kernel<<<(128 * KP + 255) / 256, 256, 0, stream>>>(
        W2, m2, W2b, 128, 784, KP, 128);
    prep_weights_kernel<<<(16 * K3 + 255) / 256, 256, 0, stream>>>(
        W3, m3, W3b, 10, 128, K3, 16);

    // Fused MLP
    const int nRows = in_sizes[0] / N1;      // 65536
    butterfly_mlp_kernel<<<nRows / MT, 256, 0, stream>>>(
        x, b1, b2, b3, W1b, W2b, W3b, out);
}
